// Unet_32504312496406
// MI455X (gfx1250) — compile-verified, matrix operand order checked by reading, not run on hardware
//
#include <hip/hip_runtime.h>
#include <hip/hip_bf16.h>

typedef _Float16 half_t;
typedef __attribute__((ext_vector_type(16))) _Float16 v16h;
typedef __attribute__((ext_vector_type(8)))  _Float16 v8h;
typedef __attribute__((ext_vector_type(8)))  float    v8f;

#define OUT_DIM 592002
#define WPAD    592008   /* f16 flat-weight per-sample stride, 16B-aligned (mult of 8 halves) */
#define BATCH   8

// ------------------------------------------------------------------
// Hypernet stage 1: h2 = (hp@W1.T + b1)@W2.T + b2   -> Hh[8][32]
// ------------------------------------------------------------------
__global__ void hyper_small_kernel(const float* __restrict__ hp,
                                   const float* __restrict__ W1,
                                   const float* __restrict__ b1,
                                   const float* __restrict__ W2,
                                   const float* __restrict__ b2,
                                   float* __restrict__ Hh) {
    int t = threadIdx.x;              // 256 threads: (b, j)
    int b = t >> 5, j = t & 31;
    float hpb = hp[b];
    float acc = b2[j];
#pragma unroll
    for (int i = 0; i < 32; ++i) {
        float h1 = hpb * W1[i] + b1[i];       // IN_DIM == 1
        acc += h1 * W2[j * 32 + i];
    }
    Hh[b * 32 + j] = acc;
}

// ------------------------------------------------------------------
// Hypernet stage 2: w = tanh(Hh @ Wo.T + bo)  (bandwidth bound on Wo)
// ------------------------------------------------------------------
__global__ void hyper_big_kernel(const float* __restrict__ Hh,
                                 const float* __restrict__ Wo,
                                 const float* __restrict__ bo,
                                 float*  __restrict__ f32w,
                                 half_t* __restrict__ f16w,
                                 float*  __restrict__ partials) {
    __shared__ float sH[256];
    __shared__ float red[8 * 256];
    int tid = threadIdx.x;
    sH[tid] = Hh[tid];
    __syncthreads();

    int j = blockIdx.x * 256 + tid;
    float labs[8];
#pragma unroll
    for (int b = 0; b < 8; ++b) labs[b] = 0.f;

    if (j < OUT_DIM) {
        float wrow[32];
#pragma unroll
        for (int k = 0; k < 32; ++k) wrow[k] = Wo[(size_t)j * 32 + k];
        float bj = bo[j];
#pragma unroll
        for (int b = 0; b < 8; ++b) {
            float acc = bj;
#pragma unroll
            for (int k = 0; k < 32; ++k) acc += sH[b * 32 + k] * wrow[k];
            float w = tanhf(acc);
            f32w[(size_t)b * OUT_DIM + j] = w;
            f16w[(size_t)b * WPAD + j] = (half_t)w;
            labs[b] = fabsf(w);
        }
    }
#pragma unroll
    for (int b = 0; b < 8; ++b) red[b * 256 + tid] = labs[b];
    __syncthreads();
    for (int s = 128; s > 0; s >>= 1) {
        if (tid < s) {
#pragma unroll
            for (int b = 0; b < 8; ++b) red[b * 256 + tid] += red[b * 256 + tid + s];
        }
        __syncthreads();
    }
    if (tid < 8) partials[(size_t)blockIdx.x * 8 + tid] = red[tid * 256];
}

// deterministic fixed-order L1 reduction
__global__ void l1_reduce_kernel(const float* __restrict__ partials, int nblk,
                                 float* __restrict__ out) {
    int b = threadIdx.x;
    if (b < 8) {
        float s = 0.f;
        for (int i = 0; i < nblk; ++i) s += partials[(size_t)i * 8 + b];
        out[b] = s;
    }
}

// zf [B,H,W,2] fp32 -> P0 [B,2,H,W] f16
__global__ void pack_zf_kernel(const float* __restrict__ zf,
                               half_t* __restrict__ dst, int HW) {
    int idx = blockIdx.x * 256 + threadIdx.x;
    if (idx >= BATCH * HW) return;
    int b = idx / HW, p = idx % HW;
    dst[((size_t)b * 2 + 0) * HW + p] = (half_t)zf[(size_t)idx * 2 + 0];
    dst[((size_t)b * 2 + 1) * HW + p] = (half_t)zf[(size_t)idx * 2 + 1];
}

// ------------------------------------------------------------------
// Implicit-GEMM 3x3 conv via v_wmma_f32_16x16x32_f16, per-sample weights.
// Block = 256 threads (8 waves), tile = 64 Cout x 32 pixels.
// Wave wv: M-tile = wv&3, N-tile = wv>>2.
// K chunked at KC = 576 (= 64 channels x 9, aligns with concat split).
// Per chunk (dynamic LDS, ~112 KB of the WGP's 320 KB):
//   s_a[64][KP]: weights staged once per block (v8h global loads)
//   s_b[32][KP]: division-free direct im2col from global
// Inner loop per wave: 2x ds_load_b128 (A) + 2x ds_load_b128 (B) + 1 WMMA.
// Rows padded to KP = KC+8 halves: 16B-aligned rows, bank-staggered.
// ------------------------------------------------------------------
template <bool KMUL32>
__global__ void __launch_bounds__(256)
conv_wmma_kernel(const half_t* __restrict__ in0, const half_t* __restrict__ in1,
                 const half_t* __restrict__ wf16, const float* __restrict__ wf32,
                 half_t* __restrict__ out,
                 int H, int W, int Cin, int Csplit, int woff, int boff, int relu) {
    extern __shared__ char smem[];
    const int tid = threadIdx.x;
    const int b  = blockIdx.z;
    const int y0 = blockIdx.y;
    const int x0 = blockIdx.x * 32;
    const int HW = H * W;
    const int Ktot = Cin * 9;
    const int KC = KMUL32 ? (Ktot > 576 ? 576 : Ktot) : 32;  // K chunk (loop extent)
    const int CC = KMUL32 ? KC / 9 : Cin;                    // channels per chunk
    const int KP = KC + 8;                                   // padded row stride

    half_t* s_a = (half_t*)smem;          // 64 x KP
    half_t* s_b = s_a + 64 * KP;          // 32 x KP

    const int lane = tid & 31;
    const int n  = lane & 15;             // N index within 16
    const int lh = (lane >> 4) & 1;       // lane-group -> +8 K offset
    const int wv = tid >> 5;              // 0..7
    const int mt = wv & 3;                // M tile
    const int nt = wv >> 2;               // N tile (0/1)
    const int mrow = (mt << 4) + n;

    v8f acc = {};
    int c0 = 0;
    for (int kc = 0; kc < Ktot; kc += KC, c0 += CC) {
        __syncthreads();                  // protect previous chunk reads
        // ---- stage A chunk: 4 threads per output-channel row ----
        {
            int m = tid >> 2;
            const half_t* wsrc = wf16 + (size_t)b * WPAD + woff + (size_t)m * Ktot + kc;
            half_t* dst = s_a + m * KP;
            if (KMUL32) {
                for (int q = (tid & 3); q < KC / 8; q += 4)
                    *(v8h*)(dst + q * 8) = *(const v8h*)(wsrc + q * 8);
            } else {                       // conv0: Ktot=18, zero-pad to 32
                for (int k = (tid & 3); k < KC; k += 4)
                    dst[k] = (k < Ktot) ? wsrc[k] : (half_t)0.f;
            }
        }
        // ---- build B chunk: direct division-free im2col from global ----
        {
            int np = tid & 31, g = tid >> 5;
            half_t* drow = s_b + np * KP;
            for (int c = g; c < CC; c += 8) {
                int cg = c0 + c;
                const half_t* src = (cg < Csplit)
                    ? in0 + ((size_t)b * Csplit + cg) * HW
                    : in1 + ((size_t)b * (Cin - Csplit) + (cg - Csplit)) * HW;
#pragma unroll
                for (int r = 0; r < 9; ++r) {
                    int yy = y0 + (r / 3) - 1;
                    int xx = x0 + np + (r % 3) - 1;
                    half_t v = (half_t)0.f;
                    if (yy >= 0 && yy < H && xx >= 0 && xx < W) v = src[yy * W + xx];
                    drow[c * 9 + r] = v;
                }
            }
            if (!KMUL32) {                 // zero pad K in [Ktot, 32)
                for (int kl = Ktot + g; kl < KC; kl += 8) drow[kl] = (half_t)0.f;
            }
        }
        __syncthreads();
        // ---- K loop: all-LDS operands, 1 WMMA per 32-K step ----
        const half_t* arow = s_a + mrow * KP + lh * 8;
        const half_t* brow = s_b + ((nt << 4) + n) * KP + lh * 8;
#pragma unroll 2
        for (int k0 = 0; k0 < KC; k0 += 32) {
            v8h alo = *(const v8h*)(arow + k0);
            v8h ahi = *(const v8h*)(arow + k0 + 16);
            v8h blo = *(const v8h*)(brow + k0);
            v8h bhi = *(const v8h*)(brow + k0 + 16);
            v16h A  = __builtin_shufflevector(alo, ahi, 0,1,2,3,4,5,6,7,8,9,10,11,12,13,14,15);
            v16h Bf = __builtin_shufflevector(blo, bhi, 0,1,2,3,4,5,6,7,8,9,10,11,12,13,14,15);
            acc = __builtin_amdgcn_wmma_f32_16x16x32_f16(false, A, false, Bf,
                                                         (short)0, acc, false, false);
        }
    }

    const int px = y0 * W + x0 + (nt << 4) + n;
#pragma unroll
    for (int r = 0; r < 8; ++r) {
        int co = (mt << 4) + r + (lh << 3);  // C/D layout: VGPR r -> M = r (+8 hi lanes)
        float v = acc[r] + wf32[(size_t)b * OUT_DIM + boff + co];
        if (relu) v = fmaxf(v, 0.f);
        out[((size_t)b * 64 + co) * HW + px] = (half_t)v;
    }
}

__global__ void maxpool_kernel(const half_t* __restrict__ in, half_t* __restrict__ out,
                               int Ho, int Wo) {
    int idx = blockIdx.x * 256 + threadIdx.x;
    int total = BATCH * 64 * Ho * Wo;
    if (idx >= total) return;
    int c = idx / (Ho * Wo);
    int rem = idx % (Ho * Wo);
    int y = rem / Wo, x = rem % Wo;
    int Wi = Wo * 2;
    const half_t* src = in + (size_t)c * (4 * Ho * Wo);
    float a0 = (float)src[(2 * y) * Wi + 2 * x];
    float a1 = (float)src[(2 * y) * Wi + 2 * x + 1];
    float a2 = (float)src[(2 * y + 1) * Wi + 2 * x];
    float a3 = (float)src[(2 * y + 1) * Wi + 2 * x + 1];
    out[idx] = (half_t)fmaxf(fmaxf(a0, a1), fmaxf(a2, a3));
}

// bilinear x2, align_corners=True
__global__ void up2_kernel(const half_t* __restrict__ in, half_t* __restrict__ out,
                           int Hi, int Wi) {
    int Ho = Hi * 2, Wo = Wi * 2;
    int idx = blockIdx.x * 256 + threadIdx.x;
    int total = BATCH * 64 * Ho * Wo;
    if (idx >= total) return;
    int c = idx / (Ho * Wo);
    int rem = idx % (Ho * Wo);
    int yo = rem / Wo, xo = rem % Wo;
    float ys = yo * (float)(Hi - 1) / (float)(Ho - 1);
    float xs = xo * (float)(Wi - 1) / (float)(Wo - 1);
    int yy0 = (int)floorf(ys); int yy1 = min(yy0 + 1, Hi - 1); float wy = ys - yy0;
    int xx0 = (int)floorf(xs); int xx1 = min(xx0 + 1, Wi - 1); float wx = xs - xx0;
    const half_t* src = in + (size_t)c * Hi * Wi;
    float v00 = src[yy0 * Wi + xx0], v01 = src[yy0 * Wi + xx1];
    float v10 = src[yy1 * Wi + xx0], v11 = src[yy1 * Wi + xx1];
    float r0 = v00 * (1.f - wy) + v10 * wy;
    float r1 = v01 * (1.f - wy) + v11 * wy;
    out[idx] = (half_t)(r0 * (1.f - wx) + r1 * wx);
}

// final 1x1 conv (64 -> 2), fp32 weights, + residual with zf; writes NHWC out
__global__ void final_kernel(const half_t* __restrict__ x, const float* __restrict__ wf32,
                             const float* __restrict__ zf, float* __restrict__ out,
                             int woff, int boff) {
    const int HW = 256 * 256;
    int idx = blockIdx.x * 256 + threadIdx.x;
    if (idx >= BATCH * HW) return;
    int b = idx / HW, p = idx % HW;
    const float* wb = wf32 + (size_t)b * OUT_DIM;
    float acc0 = wb[boff + 0];
    float acc1 = wb[boff + 1];
    const half_t* xb = x + (size_t)b * 64 * HW + p;
#pragma unroll 8
    for (int c = 0; c < 64; ++c) {
        float v = (float)xb[(size_t)c * HW];
        acc0 += v * wb[woff + c];
        acc1 += v * wb[woff + 64 + c];
    }
    out[(size_t)idx * 2 + 0] = zf[(size_t)idx * 2 + 0] + acc0;
    out[(size_t)idx * 2 + 1] = zf[(size_t)idx * 2 + 1] + acc1;
}

extern "C" void kernel_launch(void* const* d_in, const int* in_sizes, int n_in,
                              void* d_out, int out_size, void* d_ws, size_t ws_size,
                              hipStream_t stream) {
    (void)in_sizes; (void)n_in; (void)out_size; (void)ws_size;
    const float* zf = (const float*)d_in[0];
    /* d_in[1] = y (unused by the reference math) */
    const float* hp = (const float*)d_in[2];
    const float* W1 = (const float*)d_in[3];
    const float* b1 = (const float*)d_in[4];
    const float* W2 = (const float*)d_in[5];
    const float* b2 = (const float*)d_in[6];
    const float* Wo = (const float*)d_in[7];
    const float* bo = (const float*)d_in[8];
    float* out = (float*)d_out;

    char* ws = (char*)d_ws;
    size_t off = 0;
    auto alloc = [&](size_t bytes) -> void* {
        void* p = ws + off;
        off = (off + bytes + 255) & ~(size_t)255;
        return p;
    };
    const int NBLK = (OUT_DIM + 255) / 256;
    float*  Hh       = (float*)alloc((size_t)BATCH * 32 * sizeof(float));
    float*  partials = (float*)alloc((size_t)NBLK * 8 * sizeof(float));
    float*  f32w     = (float*)alloc((size_t)BATCH * OUT_DIM * sizeof(float));
    half_t* f16w     = (half_t*)alloc((size_t)BATCH * WPAD * sizeof(half_t));
    half_t* P0       = (half_t*)alloc((size_t)BATCH * 64 * 256 * 256 * sizeof(half_t));
    half_t* P1       = (half_t*)alloc((size_t)BATCH * 64 * 256 * 256 * sizeof(half_t));
    half_t* S1       = (half_t*)alloc((size_t)BATCH * 64 * 256 * 256 * sizeof(half_t));
    half_t* S2       = (half_t*)alloc((size_t)BATCH * 64 * 128 * 128 * sizeof(half_t));
    half_t* S3       = (half_t*)alloc((size_t)BATCH * 64 * 64 * 64 * sizeof(half_t));

    hyper_small_kernel<<<1, 256, 0, stream>>>(hp, W1, b1, W2, b2, Hh);
    hyper_big_kernel<<<NBLK, 256, 0, stream>>>(Hh, Wo, bo, f32w, f16w, partials);
    l1_reduce_kernel<<<1, 8, 0, stream>>>(partials, NBLK, out + (size_t)BATCH * 256 * 256 * 2);

    pack_zf_kernel<<<(BATCH * 256 * 256 + 255) / 256, 256, 0, stream>>>(zf, P0, 256 * 256);

    static const int woffs[15] = {0, 1152, 38016, 74880, 111744, 148608, 185472, 222336,
                                  259200, 332928, 369792, 443520, 480384, 554112, 590976};
    auto conv = [&](const half_t* i0, const half_t* i1, half_t* o,
                    int Hd, int Wd, int Cin, int Csplit, int li) {
        int ktot = Cin * 9;
        int kp = (ktot > 576 ? 576 : ktot) + 8;
        size_t shb = (size_t)(64 + 32) * kp * sizeof(half_t);   // s_a + s_b
        conv_wmma_kernel<true><<<dim3(Wd / 32, Hd, BATCH), 256, shb, stream>>>(
            i0, i1, f16w, f32w, o, Hd, Wd, Cin, Csplit, woffs[li], 591104 + li * 64, 1);
    };

    // conv0: Ktot=18 (guarded-K variant, KC=32 -> KP=40)
    {
        size_t shb = (size_t)(64 + 32) * 40 * sizeof(half_t);
        conv_wmma_kernel<false><<<dim3(256 / 32, 256, BATCH), 256, shb, stream>>>(
            P0, P0, f16w, f32w, P1, 256, 256, 2, 2, woffs[0], 591104, 1);
    }
    conv(P1, P1, S1, 256, 256,  64, 64, 1);
    maxpool_kernel<<<(BATCH*64*128*128 + 255)/256, 256, 0, stream>>>(S1, P0, 128, 128);
    conv(P0, P0, P1, 128, 128,  64, 64, 2);
    conv(P1, P1, S2, 128, 128,  64, 64, 3);
    maxpool_kernel<<<(BATCH*64*64*64 + 255)/256, 256, 0, stream>>>(S2, P0, 64, 64);
    conv(P0, P0, P1,  64,  64,  64, 64, 4);
    conv(P1, P1, S3,  64,  64,  64, 64, 5);
    maxpool_kernel<<<(BATCH*64*32*32 + 255)/256, 256, 0, stream>>>(S3, P0, 32, 32);
    conv(P0, P0, P1,  32,  32,  64, 64, 6);
    conv(P1, P1, P0,  32,  32,  64, 64, 7);
    // decoder
    up2_kernel<<<(BATCH*64*64*64 + 255)/256, 256, 0, stream>>>(P0, P1, 32, 32);
    conv(P1, S3, P0,  64,  64, 128, 64, 8);
    conv(P0, P0, P1,  64,  64,  64, 64, 9);
    up2_kernel<<<(BATCH*64*128*128 + 255)/256, 256, 0, stream>>>(P1, P0, 64, 64);
    conv(P0, S2, P1, 128, 128, 128, 64, 10);
    conv(P1, P1, P0, 128, 128,  64, 64, 11);
    up2_kernel<<<(BATCH*64*256*256 + 255)/256, 256, 0, stream>>>(P0, P1, 128, 128);
    conv(P1, S1, P0, 256, 256, 128, 64, 12);
    conv(P0, P0, P1, 256, 256,  64, 64, 13);

    final_kernel<<<(BATCH * 256 * 256 + 255) / 256, 256, 0, stream>>>(
        P1, f32w, zf, out, 590976, 592000);
}
